// AdaptiveExpertRouter_67705864454660
// MI455X (gfx1250) — matrix operand order repaired and checked by reading, not compiled, not run on hardware
//
#include <hip/hip_runtime.h>
#include <hip/hip_bf16.h>
#include <math.h>

// ---------------------------------------------------------------------------
// AdaptiveExpertRouter for MI455X (gfx1250, wave32, WMMA)
//   Heavy GEMMs in bf16 via v_wmma_f32_16x16x32_bf16 (f32 accumulate),
//   64x64 output tile per wave (4x4 fragments -> 16 WMMA per 16 B128 loads).
//   LN / GELU / softmax / top-2 / routed combine in fp32 (bandwidth bound).
// ---------------------------------------------------------------------------

typedef __attribute__((ext_vector_type(16))) __bf16 v16bf;
typedef __attribute__((ext_vector_type(8)))  float  v8f;

#define SDIM 1024
#define TDIM 4096
#define NEXP 8

__device__ __forceinline__ float gelu_exact(float x) {
    return 0.5f * x * (1.0f + erff(x * 0.70710678118654752f));
}

// ---------------------------------------------------------------------------
// WMMA fragment loaders (16-bit 16x32 A, 32x16 B layouts from CDNA5 ISA 7.12.2)
//  A: lane l holds row m = m0 + (l&15); lanes 0-15 carry K = kb+0..7 & kb+16..23,
//     lanes 16-31 carry K = kb+8..15 & kb+24..31  -> two 16B loads per lane.
//  B: lane l holds col n = n0 + (l&15); lanes 0-15 carry K = kb+0..7 & kb+16..23,
//     lanes 16-31 carry K = kb+8..15 & kb+24..31             -> two 16B loads per lane.
// Both A (row-major [M,K]) and W (row-major [Nc,K]) are K-contiguous in memory.
// ---------------------------------------------------------------------------
__device__ __forceinline__ v16bf load_a_frag(const __bf16* __restrict__ A, int lda,
                                             int m0, int kb, int lane) {
    const int hi = lane >> 4;
    const __bf16* base = A + (size_t)(m0 + (lane & 15)) * lda + kb + 8 * hi;
    union { v16bf v; uint4 q[2]; } u;
    u.q[0] = *reinterpret_cast<const uint4*>(base);       // K = kb+8*hi   .. +7
    u.q[1] = *reinterpret_cast<const uint4*>(base + 16);  // K = kb+16+8*hi.. +7
    return u.v;
}

__device__ __forceinline__ v16bf load_b_frag(const __bf16* __restrict__ W, int ldw,
                                             int n0, int kb, int lane) {
    const int hi = lane >> 4;
    const __bf16* base = W + (size_t)(n0 + (lane & 15)) * ldw + kb + 8 * hi;
    union { v16bf v; uint4 q[2]; } u;
    u.q[0] = *reinterpret_cast<const uint4*>(base);       // K = kb+8*hi   .. +7
    u.q[1] = *reinterpret_cast<const uint4*>(base + 16);  // K = kb+16+8*hi.. +7
    return u.v;
}

// ---------------------------------------------------------------------------
// GEMM: C[M,Nc] (fp32) = A[M,K](bf16) @ W[Nc,K](bf16)^T + bias
// Block = 128 threads = 4 waves (2x2); block tile 128x128; wave tile 64x64
// (4x4 WMMA fragments = 16 v_wmma per 32-deep K step).
// Grid: (Nc/128, M/128). M,Nc multiples of 128; K multiple of 32.
// ---------------------------------------------------------------------------
__global__ __launch_bounds__(128)
void gemm_bf16_kernel(const __bf16* __restrict__ A, int lda,
                      const __bf16* __restrict__ W, int ldw,
                      const float* __restrict__ bias,
                      float* __restrict__ C, int ldc, int K)
{
    const int lane = threadIdx.x & 31;
    const int wv   = threadIdx.x >> 5;
    const int m0 = blockIdx.y * 128 + (wv & 1) * 64;
    const int n0 = blockIdx.x * 128 + (wv >> 1) * 64;

    v8f acc[4][4] = {};

    for (int kb = 0; kb < K; kb += 32) {
        v16bf a[4];
        v16bf b[4];
#pragma unroll
        for (int i = 0; i < 4; ++i) a[i] = load_a_frag(A, lda, m0 + 16 * i, kb, lane);
#pragma unroll
        for (int j = 0; j < 4; ++j) b[j] = load_b_frag(W, ldw, n0 + 16 * j, kb, lane);

        // Speculative prefetch of the tile after next into WGP-level caches.
        if (kb + 64 < K) {
            __builtin_prefetch(A + (size_t)(m0 + (lane & 15)) * lda + kb + 64, 0, 3);
            __builtin_prefetch(W + (size_t)(n0 + (lane & 15)) * ldw + kb + 64, 0, 3);
        }

#pragma unroll
        for (int i = 0; i < 4; ++i)
#pragma unroll
            for (int j = 0; j < 4; ++j)
                acc[i][j] = __builtin_amdgcn_wmma_f32_16x16x32_bf16(
                    false, a[i], false, b[j], (short)0, acc[i][j], false, false);
    }

    // Epilogue: D fragment layout -> lane (l&15) = column, VGPR r -> row r + 8*(l>>4)
    const int col = lane & 15;
    const int hi  = lane >> 4;
#pragma unroll
    for (int i = 0; i < 4; ++i) {
#pragma unroll
        for (int j = 0; j < 4; ++j) {
            const int n = n0 + 16 * j + col;
            const float bv = bias ? bias[n] : 0.0f;
#pragma unroll
            for (int r = 0; r < 8; ++r) {
                const int m = m0 + 16 * i + 8 * hi + r;
                C[(size_t)m * ldc + n] = acc[i][j][r] + bv;
            }
        }
    }
}

// ---------------------------------------------------------------------------
// Row-wise post-processing: mode 0 = convert f32->bf16
//                           mode 1 = GELU -> bf16
//                           mode 2 = LayerNorm (gamma,beta) -> GELU -> bf16
// One 256-thread block per row.
// ---------------------------------------------------------------------------
__global__ __launch_bounds__(256)
void postproc_kernel(const float* __restrict__ X, int ldx,
                     const float* __restrict__ gamma, const float* __restrict__ beta,
                     __bf16* __restrict__ Y, int ldy,
                     int cols, int mode)
{
    const int row = blockIdx.x;
    const int tid = threadIdx.x;
    const float* x = X + (size_t)row * ldx;
    __bf16* y = Y + (size_t)row * ldy;

    if (mode == 2) {
        float s = 0.0f, ss = 0.0f;
        for (int j = tid; j < cols; j += 256) {
            float v = x[j];
            s  += v;
            ss += v * v;
        }
        __shared__ float rs[256];
        __shared__ float rq[256];
        rs[tid] = s; rq[tid] = ss;
        __syncthreads();
        for (int st = 128; st > 0; st >>= 1) {
            if (tid < st) { rs[tid] += rs[tid + st]; rq[tid] += rq[tid + st]; }
            __syncthreads();
        }
        const float inv_n = 1.0f / (float)cols;
        const float mu  = rs[0] * inv_n;
        const float var = rq[0] * inv_n - mu * mu;
        const float rinv = rsqrtf(var + 1e-5f);
        for (int j = tid; j < cols; j += 256) {
            float v = (x[j] - mu) * rinv * gamma[j] + beta[j];
            y[j] = (__bf16)gelu_exact(v);
        }
    } else if (mode == 1) {
        for (int j = tid; j < cols; j += 256)
            y[j] = (__bf16)gelu_exact(x[j]);
    } else {
        for (int j = tid; j < cols; j += 256)
            y[j] = (__bf16)x[j];
    }
}

// ---------------------------------------------------------------------------
// Small logits: L[t,e] = sum_k X_bf16[t,k] * Wt[e,k] + b[e]   (E = 8)
// One 256-thread block per token.
// ---------------------------------------------------------------------------
__global__ __launch_bounds__(256)
void logits_kernel(const __bf16* __restrict__ X, int ldx,
                   const float* __restrict__ Wt, const float* __restrict__ b,
                   float* __restrict__ L, int K)
{
    const int t   = blockIdx.x;
    const int tid = threadIdx.x;
    const __bf16* x = X + (size_t)t * ldx;

    float acc[NEXP];
#pragma unroll
    for (int e = 0; e < NEXP; ++e) acc[e] = 0.0f;

    for (int k = tid; k < K; k += 256) {
        const float xv = (float)x[k];
#pragma unroll
        for (int e = 0; e < NEXP; ++e)
            acc[e] += xv * Wt[(size_t)e * K + k];
    }

    __shared__ float red[256 * NEXP];
#pragma unroll
    for (int e = 0; e < NEXP; ++e) red[tid * NEXP + e] = acc[e];
    __syncthreads();
    for (int st = 128; st > 0; st >>= 1) {
        if (tid < st) {
#pragma unroll
            for (int e = 0; e < NEXP; ++e)
                red[tid * NEXP + e] += red[(tid + st) * NEXP + e];
        }
        __syncthreads();
    }
    if (tid < NEXP)
        L[(size_t)t * NEXP + tid] = red[tid] + b[tid];
}

// ---------------------------------------------------------------------------
// Routing: softmax(cap/temp), softmax(gap), softmax(es); 0.4/0.3/0.3 combine;
// top-2 (first-occurrence tie-break like lax.top_k); renormalize; weighted
// combine of the two selected expert rows. One 256-thread block per token.
// ---------------------------------------------------------------------------
__device__ __forceinline__ void softmax8(float* v) {
    float m = v[0];
#pragma unroll
    for (int i = 1; i < NEXP; ++i) m = fmaxf(m, v[i]);
    float s = 0.0f;
#pragma unroll
    for (int i = 0; i < NEXP; ++i) { v[i] = expf(v[i] - m); s += v[i]; }
    const float r = 1.0f / s;
#pragma unroll
    for (int i = 0; i < NEXP; ++i) v[i] *= r;
}

__global__ __launch_bounds__(256)
void routing_kernel(const float* __restrict__ capL, const float* __restrict__ gapL,
                    const float* __restrict__ esL, const float* __restrict__ temp,
                    const float* __restrict__ EO,   // [E, Ntok, TDIM]
                    float* __restrict__ out, int Ntok)
{
    const int t = blockIdx.x;
    const float tinv = 1.0f / temp[0];

    float sc[NEXP], sg[NEXP], se[NEXP];
#pragma unroll
    for (int e = 0; e < NEXP; ++e) {
        sc[e] = capL[(size_t)t * NEXP + e] * tinv;
        sg[e] = gapL[(size_t)t * NEXP + e];
        se[e] = esL[(size_t)t * NEXP + e];
    }
    softmax8(sc); softmax8(sg); softmax8(se);

    float comb[NEXP];
#pragma unroll
    for (int e = 0; e < NEXP; ++e)
        comb[e] = 0.4f * sc[e] + 0.3f * sg[e] + 0.3f * se[e];

    int i1 = 0; float s1 = comb[0];
#pragma unroll
    for (int e = 1; e < NEXP; ++e)
        if (comb[e] > s1) { s1 = comb[e]; i1 = e; }
    int i2 = (i1 == 0) ? 1 : 0; float s2 = comb[i2];
#pragma unroll
    for (int e = 0; e < NEXP; ++e)
        if (e != i1 && comb[e] > s2) { s2 = comb[e]; i2 = e; }

    const float denom = s1 + s2 + 1e-8f;
    const float w1 = s1 / denom;
    const float w2 = s2 / denom;

    const float* e1 = EO + ((size_t)i1 * Ntok + t) * TDIM;
    const float* e2 = EO + ((size_t)i2 * Ntok + t) * TDIM;
    float* o = out + (size_t)t * TDIM;
    for (int j = threadIdx.x; j < TDIM; j += 256)
        o[j] = w1 * e1[j] + w2 * e2[j];
}

// ---------------------------------------------------------------------------
// Host-side orchestration
// ---------------------------------------------------------------------------
extern "C" void kernel_launch(void* const* d_in, const int* in_sizes, int n_in,
                              void* d_out, int out_size, void* d_ws, size_t ws_size,
                              hipStream_t stream) {
    (void)n_in; (void)out_size; (void)ws_size;

    const int Ntok = in_sizes[0] / SDIM;   // B*S = 4096

    // ---- inputs (setup_inputs dict order, params flattened in insertion order)
    const float* student = (const float*)d_in[0];   // [Ntok, SDIM]
    const float* EO      = (const float*)d_in[1];   // [E, Ntok, TDIM]
    const float* teacher = EO;                      // expert 0
    const float* cap_w1  = (const float*)d_in[2];   // [2S, S]
    const float* cap_b1  = (const float*)d_in[3];
    const float* cap_g1  = (const float*)d_in[4];
    const float* cap_be1 = (const float*)d_in[5];
    const float* cap_w2  = (const float*)d_in[6];   // [S, 2S]
    const float* cap_b2  = (const float*)d_in[7];
    const float* cap_g2  = (const float*)d_in[8];
    const float* cap_be2 = (const float*)d_in[9];
    const float* cap_w3  = (const float*)d_in[10];  // [E, S]
    const float* cap_b3  = (const float*)d_in[11];
    const float* temp    = (const float*)d_in[12];  // [1]
    const float* t2s_w   = (const float*)d_in[13];  // [S, T]
    const float* t2s_b   = (const float*)d_in[14];
    const float* gap_w1  = (const float*)d_in[15];  // [T, 2S]
    const float* gap_b1  = (const float*)d_in[16];
    const float* gap_g1  = (const float*)d_in[17];
    const float* gap_be1 = (const float*)d_in[18];
    const float* gap_w2  = (const float*)d_in[19];  // [E, T]
    const float* gap_b2  = (const float*)d_in[20];
    const float* es_w1   = (const float*)d_in[21];  // [T/2, T]
    const float* es_b1   = (const float*)d_in[22];
    const float* es_w2   = (const float*)d_in[23];  // [E, T/2]
    const float* es_b2   = (const float*)d_in[24];

    float* out = (float*)d_out;                     // [Ntok, TDIM]

    // ---- workspace carve-up (256B aligned)
    char* wsb = (char*)d_ws;
    size_t off = 0;
    auto alloc = [&](size_t bytes) -> void* {
        void* p = wsb + off;
        off = (off + bytes + 255) & ~(size_t)255;
        return p;
    };
    __bf16* concat  = (__bf16*)alloc((size_t)Ntok * 2048 * 2);  // [student | t2s proj]
    __bf16* teachB  = (__bf16*)alloc((size_t)Ntok * TDIM * 2);
    __bf16* wcap1b  = (__bf16*)alloc((size_t)2048 * 1024 * 2);
    __bf16* wcap2b  = (__bf16*)alloc((size_t)1024 * 2048 * 2);
    __bf16* wt2sb   = (__bf16*)alloc((size_t)1024 * 4096 * 2);
    __bf16* wgap1b  = (__bf16*)alloc((size_t)4096 * 2048 * 2);
    __bf16* wes1b   = (__bf16*)alloc((size_t)2048 * 4096 * 2);
    __bf16* h1      = (__bf16*)alloc((size_t)Ntok * 2048 * 2);
    __bf16* h2      = (__bf16*)alloc((size_t)Ntok * 1024 * 2);
    __bf16* gact    = (__bf16*)alloc((size_t)Ntok * 4096 * 2);
    __bf16* eact    = (__bf16*)alloc((size_t)Ntok * 2048 * 2);
    float*  gemmC   = (float*)alloc((size_t)Ntok * 4096 * 4);   // reused fp32 GEMM out
    float*  capL    = (float*)alloc((size_t)Ntok * NEXP * 4);
    float*  gapL    = (float*)alloc((size_t)Ntok * NEXP * 4);
    float*  esL     = (float*)alloc((size_t)Ntok * NEXP * 4);

    const dim3 b256(256), b128(128);

    // ---- precision conversion (fp32 -> bf16)
    postproc_kernel<<<Ntok, b256, 0, stream>>>(student, SDIM, nullptr, nullptr,
                                               concat, 2048, SDIM, 0);
    postproc_kernel<<<Ntok, b256, 0, stream>>>(teacher, TDIM, nullptr, nullptr,
                                               teachB, TDIM, TDIM, 0);
    postproc_kernel<<<2048, b256, 0, stream>>>(cap_w1, 1024, nullptr, nullptr, wcap1b, 1024, 1024, 0);
    postproc_kernel<<<1024, b256, 0, stream>>>(cap_w2, 2048, nullptr, nullptr, wcap2b, 2048, 2048, 0);
    postproc_kernel<<<1024, b256, 0, stream>>>(t2s_w,  4096, nullptr, nullptr, wt2sb,  4096, 4096, 0);
    postproc_kernel<<<4096, b256, 0, stream>>>(gap_w1, 2048, nullptr, nullptr, wgap1b, 2048, 2048, 0);
    postproc_kernel<<<2048, b256, 0, stream>>>(es_w1,  4096, nullptr, nullptr, wes1b,  4096, 4096, 0);

    // ---- capacity net: Lin -> LN -> GELU -> Lin -> LN -> GELU -> Lin(8)
    gemm_bf16_kernel<<<dim3(2048 / 128, Ntok / 128), b128, 0, stream>>>(
        concat, 2048, wcap1b, 1024, cap_b1, gemmC, 2048, 1024);
    postproc_kernel<<<Ntok, b256, 0, stream>>>(gemmC, 2048, cap_g1, cap_be1, h1, 2048, 2048, 2);

    gemm_bf16_kernel<<<dim3(1024 / 128, Ntok / 128), b128, 0, stream>>>(
        h1, 2048, wcap2b, 2048, cap_b2, gemmC, 1024, 2048);
    postproc_kernel<<<Ntok, b256, 0, stream>>>(gemmC, 1024, cap_g2, cap_be2, h2, 1024, 1024, 2);

    logits_kernel<<<Ntok, b256, 0, stream>>>(h2, 1024, cap_w3, cap_b3, capL, 1024);

    // ---- gap analyzer: teacher_proj -> concat -> Lin -> LN -> GELU -> Lin(8)
    gemm_bf16_kernel<<<dim3(1024 / 128, Ntok / 128), b128, 0, stream>>>(
        teachB, 4096, wt2sb, 4096, t2s_b, gemmC, 1024, 4096);
    postproc_kernel<<<Ntok, b256, 0, stream>>>(gemmC, 1024, nullptr, nullptr,
                                               concat + 1024, 2048, 1024, 0);

    gemm_bf16_kernel<<<dim3(4096 / 128, Ntok / 128), b128, 0, stream>>>(
        concat, 2048, wgap1b, 2048, gap_b1, gemmC, 4096, 2048);
    postproc_kernel<<<Ntok, b256, 0, stream>>>(gemmC, 4096, gap_g1, gap_be1, gact, 4096, 4096, 2);

    logits_kernel<<<Ntok, b256, 0, stream>>>(gact, 4096, gap_w2, gap_b2, gapL, 4096);

    // ---- expert scorer: Lin -> GELU -> Lin(8)
    gemm_bf16_kernel<<<dim3(2048 / 128, Ntok / 128), b128, 0, stream>>>(
        teachB, 4096, wes1b, 4096, es_b1, gemmC, 2048, 4096);
    postproc_kernel<<<Ntok, b256, 0, stream>>>(gemmC, 2048, nullptr, nullptr, eact, 2048, 2048, 1);

    logits_kernel<<<Ntok, b256, 0, stream>>>(eact, 2048, es_w2, es_b2, esL, 2048);

    // ---- combine, top-2 route, weighted expert mix
    routing_kernel<<<Ntok, b256, 0, stream>>>(capL, gapL, esL, temp, EO, out, Ntok);
}
